// QuanvolutionHybrid_65481071406826
// MI455X (gfx1250) — matrix operand order repaired and checked by reading, not compiled, not run on hardware
//
#include <hip/hip_runtime.h>
#include <hip/hip_bf16.h>

typedef __attribute__((ext_vector_type(16))) _Float16 v16h;
typedef __attribute__((ext_vector_type(8)))  _Float16 v8h;
typedef __attribute__((ext_vector_type(8)))  float    v8f;

#define BATCH 4096

// ---- CDNA5 async global->LDS (ASYNCcnt) with compile-safe fallback --------
typedef int gv4i __attribute__((vector_size(16)));        // matches builtin sig

#if __has_builtin(__builtin_amdgcn_global_load_async_to_lds_b128)
#define HAVE_ASYNC_B128 1
#endif
#if __has_builtin(__builtin_amdgcn_global_load_async_to_lds_b32)
#define HAVE_ASYNC_B32 1
#endif

#define G1I(p)  ((__attribute__((address_space(1))) int*)(p))
#define L3I(p)  ((__attribute__((address_space(3))) int*)(p))
#define G1V(p)  ((__attribute__((address_space(1))) gv4i*)(p))
#define L3V(p)  ((__attribute__((address_space(3))) gv4i*)(p))

__device__ __forceinline__ void async_wait0() {
#if __has_builtin(__builtin_amdgcn_s_wait_asynccnt)
  __builtin_amdgcn_s_wait_asynccnt(0);
#else
  asm volatile("s_wait_asynccnt 0" ::: "memory");
#endif
}

__device__ __forceinline__ v16h cat8(v8h lo, v8h up) {
  return __builtin_shufflevector(lo, up, 0,1,2,3,4,5,6,7,8,9,10,11,12,13,14,15);
}

// ---------------------------------------------------------------------------
// Gate tables (structural stand-ins for _make_ops(1,8) / _make_ops(2,50)).
// kind: 0=RX 1=RY 2=RZ 3=CRX(w0=control,w1=target). param index == op index.
// constexpr + full unroll keeps the 16-amplitude state entirely in VGPRs.
// ---------------------------------------------------------------------------
constexpr int ENC_K[8]  = {1,0,3,2,1,3,0,2};
constexpr int ENC_W0[8] = {0,2,1,3,2,0,1,2};
constexpr int ENC_W1[8] = {0,0,3,0,0,2,0,0};

constexpr int QFC_K[50] = {0,1,3,2,0,3,1,2,0,1, 3,2,0,1,3,2,0,3,1,2,
                           0,3,1,2,0,3,1,2,0,3, 1,2,0,3,1,2,0,3,1,2,
                           0,3,1,2,0,3,1,2,0,3};
constexpr int QFC_W0[50]= {2,0,1,3,1,0,2,0,3,1, 2,2,0,3,3,1,2,1,0,3,
                           1,0,2,0,3,2,1,2,0,3, 3,1,2,1,0,3,1,0,2,0,
                           3,2,1,2,0,3,3,1,2,1};
constexpr int QFC_W1[50]= {0,0,3,0,0,2,0,0,0,0, 1,0,0,0,0,0,0,0,0,0,
                           0,1,0,0,0,3,0,0,0,2, 0,0,0,2,0,0,0,3,0,0,
                           0,0,0,0,0,1,0,0,0,3};

// ---------------- quantum gate engine (register-resident) -----------------
__device__ __forceinline__ void rot1(float* sr, float* si, int st,
    float u00r, float u00i, float u01r, float u01i,
    float u10r, float u10i, float u11r, float u11i) {
  #pragma unroll
  for (int i = 0; i < 16; ++i) {
    if ((i & st) == 0) {
      const int j = i | st;
      float ar = sr[i], ai = si[i], br = sr[j], bi = si[j];
      sr[i] = u00r*ar - u00i*ai + u01r*br - u01i*bi;
      si[i] = u00r*ai + u00i*ar + u01r*bi + u01i*br;
      sr[j] = u10r*ar - u10i*ai + u11r*br - u11i*bi;
      si[j] = u10r*ai + u10i*ar + u11r*bi + u11i*br;
    }
  }
}

__device__ __forceinline__ void crot1(float* sr, float* si, int sc, int st,
    float u00r, float u00i, float u01r, float u01i,
    float u10r, float u10i, float u11r, float u11i) {
  #pragma unroll
  for (int i = 0; i < 16; ++i) {
    if ((i & sc) != 0 && (i & st) == 0) {
      const int j = i | st;
      float ar = sr[i], ai = si[i], br = sr[j], bi = si[j];
      sr[i] = u00r*ar - u00i*ai + u01r*br - u01i*bi;
      si[i] = u00r*ai + u00i*ar + u01r*bi + u01i*br;
      sr[j] = u10r*ar - u10i*ai + u11r*br - u11i*bi;
      si[j] = u10r*ai + u10i*ar + u11r*bi + u11i*br;
    }
  }
}

__device__ __forceinline__ void do_gate(float* sr, float* si,
                                        int kind, int w0, int w1, float th) {
  const float c = __cosf(0.5f*th), s = __sinf(0.5f*th);
  if (kind == 0)      rot1 (sr, si, 8 >> w0,  c,0.f,  0.f,-s,  0.f,-s,  c,0.f); // RX
  else if (kind == 1) rot1 (sr, si, 8 >> w0,  c,0.f,  -s,0.f,  s,0.f,   c,0.f); // RY
  else if (kind == 2) rot1 (sr, si, 8 >> w0,  c,-s,   0.f,0.f, 0.f,0.f, c,s);   // RZ
  else                crot1(sr, si, 8 >> w0, 8 >> w1,
                            c,0.f, 0.f,-s, 0.f,-s, c,0.f);                       // CRX
}

__device__ __forceinline__ void measure_z(const float* sr, const float* si, float* e) {
  #pragma unroll
  for (int w = 0; w < 4; ++w) e[w] = 0.f;
  #pragma unroll
  for (int i = 0; i < 16; ++i) {
    const float p = sr[i]*sr[i] + si[i]*si[i];
    #pragma unroll
    for (int w = 0; w < 4; ++w) e[w] += (i & (8 >> w)) ? -p : p;
  }
}

// ---------------- K1: conv1 + relu + 2x2 maxpool -> h1 f16 NHWC ------------
__global__ __launch_bounds__(256) void k_conv1(
    const float* __restrict__ x, const float* __restrict__ w,
    const float* __restrict__ bias, _Float16* __restrict__ h1) {
  __shared__ __align__(16) float sxp[1156];                 // 34x34 zero-padded
  const int b  = blockIdx.x;
  const int co = blockIdx.y;
  const int t  = threadIdx.x;
  {
    const float4 z = {0.f, 0.f, 0.f, 0.f};
    for (int j = t; j < 289; j += 256) reinterpret_cast<float4*>(sxp)[j] = z;
  }
  float wv[9];
  #pragma unroll
  for (int i = 0; i < 9; ++i) wv[i] = w[co*9 + i];
  const float bv = bias[co];
  __syncthreads();
  #pragma unroll
  for (int u = 0; u < 4; ++u) {                             // interior copy
    const int j = t + u*256;
    const int y = j >> 5, xx = j & 31;
#if defined(HAVE_ASYNC_B32)
    __builtin_amdgcn_global_load_async_to_lds_b32(
        G1I(x + (size_t)b*1024 + j), L3I(&sxp[(y + 1)*34 + xx + 1]), 0, 0);
#else
    sxp[(y + 1)*34 + xx + 1] = x[(size_t)b*1024 + j];
#endif
  }
#if defined(HAVE_ASYNC_B32)
  async_wait0();
#endif
  __syncthreads();
  const int py = t >> 4, px = t & 15;
  float m = 0.f;                                            // relu values >= 0
  #pragma unroll
  for (int dy = 0; dy < 2; ++dy)
  #pragma unroll
  for (int dx = 0; dx < 2; ++dx) {
    const int cy = 2*py + dy, cx = 2*px + dx;
    float s = bv;
    #pragma unroll
    for (int i = 0; i < 3; ++i)
      #pragma unroll
      for (int j = 0; j < 3; ++j)
        s = fmaf(sxp[(cy + i)*34 + cx + j], wv[i*3 + j], s);
    m = fmaxf(m, fmaxf(s, 0.f));
  }
  h1[((size_t)b*256 + t)*8 + co] = (_Float16)m;
}

// ---- K2: conv2 as implicit GEMM on WMMA + relu + pool + channel mean ------
// Per image: M=256 pixels, N=16 channels, K=72 (pad) -> 16 tiles x 3 k-steps.
// im2col built in LDS with compile-time k mapping; fragments = ds_load_b128.
__global__ __launch_bounds__(128) void k_conv2_wmma(
    const _Float16* __restrict__ h1, const float* __restrict__ w2,
    const float* __restrict__ b2, float* __restrict__ patches) {
  __shared__ __align__(16) _Float16 sh1p[2592];             // 18x18x8 padded
  __shared__ __align__(16) _Float16 sw[1536];               // [N=16][K=96]
  __shared__ __align__(16) _Float16 sA[20480];              // [256 pix][K=80]
  __shared__ _Float16 sout[4096];                           // 256 pix x 16 co
  const int b = blockIdx.x;
  const int t = threadIdx.x;
  const int lane = t & 31, wave = t >> 5;
  if (b + 1 < BATCH)                                        // global_prefetch_b8
    __builtin_prefetch(h1 + (size_t)(b + 1)*2048, 0, 1);
  {                                                         // zero padded tile
    const uint4 z = {0u, 0u, 0u, 0u};
    for (int j = t; j < 324; j += 128) reinterpret_cast<uint4*>(sh1p)[j] = z;
  }
  __syncthreads();
  {                                                         // interior copy
    const int y = t >> 3, s = t & 7;                        // 16 rows x 8 segs
#if defined(HAVE_ASYNC_B128)
    __builtin_amdgcn_global_load_async_to_lds_b128(
        G1V(h1 + (size_t)b*2048 + y*128 + s*8),
        L3V(&sh1p[((y + 1)*18 + 1)*8 + s*8]), 0, 0);
#else
    reinterpret_cast<uint4*>(&sh1p[((y + 1)*18 + 1)*8])[s] =
        reinterpret_cast<const uint4*>(h1 + (size_t)b*2048 + y*128)[s];
#endif
  }
  {                                                         // weights [n][96]
    const int n = t & 15, seg = t >> 4;
    #pragma unroll
    for (int u = 0; u < 12; ++u) {
      const int k = seg*12 + u;
      const int kc = (k < 72) ? k : 0;
      const _Float16 v = (_Float16)w2[n*72 + kc];
      sw[n*96 + k] = (k < 72) ? v : (_Float16)0.f;
    }
  }
#if defined(HAVE_ASYNC_B128)
  async_wait0();
#endif
  __syncthreads();
  #pragma unroll
  for (int pp = 0; pp < 2; ++pp) {                          // im2col -> sA
    const int pix = t + pp*128;
    const int y = pix >> 4, x = pix & 15;
    v8h nb[9];
    #pragma unroll
    for (int r = 0; r < 9; ++r)                             // 9 taps x 8 ci
      nb[r] = *(const v8h*)&sh1p[((y + r/3)*18 + (x + r%3))*8];
    #pragma unroll
    for (int g = 0; g < 10; ++g) {                          // K groups of 8
      v8h pk = {};
      #pragma unroll
      for (int e = 0; e < 8; ++e) {
        const int k = g*8 + e;                              // compile-time
        if (k < 72) pk[e] = nb[k % 9][k / 9];               // k = ci*9+ky*3+kx
      }
      *(v8h*)&sA[pix*80 + g*8] = pk;
    }
  }
  __syncthreads();
  const int hi = (lane >> 4) & 1;
  const int n  = lane & 15;
  const _Float16* bp = &sw[n*96];
  const v16h bf0 = cat8(*(const v8h*)&bp[      8*hi], *(const v8h*)&bp[16 + 8*hi]);
  const v16h bf1 = cat8(*(const v8h*)&bp[32 + 8*hi], *(const v8h*)&bp[48 + 8*hi]);
  const v16h bf2 = cat8(*(const v8h*)&bp[64 + 8*hi], *(const v8h*)&bp[80 + 8*hi]);
  const float bias = b2[n];
  const v8h z8 = {};
  #pragma unroll
  for (int tl = 0; tl < 4; ++tl) {
    const int T = wave*4 + tl;
    const _Float16* ap = &sA[(T*16 + (lane & 15))*80];      // A row = lane%16
    const v16h a0 = cat8(*(const v8h*)&ap[     8*hi], *(const v8h*)&ap[16 + 8*hi]);
    const v16h a1 = cat8(*(const v8h*)&ap[32 + 8*hi], *(const v8h*)&ap[48 + 8*hi]);
    const v16h a2 = cat8(*(const v8h*)&ap[64 + 8*hi], z8);  // K 80..95 == 0
    v8f c = {};
    c = __builtin_amdgcn_wmma_f32_16x16x32_f16(false, a0, false, bf0, (short)0, c, false, false);
    c = __builtin_amdgcn_wmma_f32_16x16x32_f16(false, a1, false, bf1, (short)0, c, false, false);
    c = __builtin_amdgcn_wmma_f32_16x16x32_f16(false, a2, false, bf2, (short)0, c, false, false);
    #pragma unroll
    for (int r = 0; r < 8; ++r) {                           // D row = r + 8*hi
      const int p2 = T*16 + r + 8*hi;
      sout[p2*16 + n] = (_Float16)fmaxf(c[r] + bias, 0.f);
    }
  }
  __syncthreads();
  if (t < 64) {                                             // pool + chan-mean
    const int pr = t >> 3, pc = t & 7;
    float sum = 0.f;
    #pragma unroll
    for (int co = 0; co < 16; ++co) {
      const float v0 = (float)sout[((2*pr  )*16 + 2*pc  )*16 + co];
      const float v1 = (float)sout[((2*pr  )*16 + 2*pc+1)*16 + co];
      const float v2 = (float)sout[((2*pr+1)*16 + 2*pc  )*16 + co];
      const float v3 = (float)sout[((2*pr+1)*16 + 2*pc+1)*16 + co];
      sum += fmaxf(fmaxf(v0, v1), fmaxf(v2, v3));
    }
    const int p  = (pr >> 1)*4 + (pc >> 1);
    const int ii = (pr & 1)*2 + (pc & 1);
    patches[(size_t)b*64 + p*4 + ii] = sum * (1.f/16.f);
  }
}

// ---------------- K3: per-patch encoder circuit + <Z> ----------------------
__global__ __launch_bounds__(256) void k_encoder(
    const float* __restrict__ patches, const float* __restrict__ enc_p,
    float* __restrict__ feats, float* __restrict__ st15) {
  const int tid = blockIdx.x*blockDim.x + threadIdx.x;      // 65536
  const int b = tid >> 4, p = tid & 15;
  float cw[4], swv[4];
  #pragma unroll
  for (int w = 0; w < 4; ++w) {
    const float th = patches[(size_t)b*64 + p*4 + w];
    cw[w] = __cosf(0.5f*th); swv[w] = __sinf(0.5f*th);
  }
  float sr[16], si[16];                                     // RY-encode |0000>
  #pragma unroll
  for (int i = 0; i < 16; ++i) {
    float a = (i & 8) ? swv[0] : cw[0];
    a *= (i & 4) ? swv[1] : cw[1];
    a *= (i & 2) ? swv[2] : cw[2];
    a *= (i & 1) ? swv[3] : cw[3];
    sr[i] = a; si[i] = 0.f;
  }
  #pragma unroll
  for (int g = 0; g < 8; ++g)
    do_gate(sr, si, ENC_K[g], ENC_W0[g], ENC_W1[g], enc_p[g]);
  float e[4]; measure_z(sr, si, e);
  #pragma unroll
  for (int w = 0; w < 4; ++w) feats[(size_t)b*68 + p*4 + w] = e[w];
  if (p == 15) {
    #pragma unroll
    for (int i = 0; i < 16; ++i) {
      st15[(size_t)b*32 + i]      = sr[i];
      st15[(size_t)b*32 + 16 + i] = si[i];
    }
  }
}

// ---------------- K4: QFC circuit on patch-15 state ------------------------
__global__ __launch_bounds__(256) void k_qfc(
    const float* __restrict__ st15, const float* __restrict__ rp,
    const float* __restrict__ qp, float* __restrict__ q) {
  const int b = blockIdx.x*blockDim.x + threadIdx.x;        // 4096
  float sr[16], si[16];
  #pragma unroll
  for (int i = 0; i < 16; ++i) {
    sr[i] = st15[(size_t)b*32 + i];
    si[i] = st15[(size_t)b*32 + 16 + i];
  }
  #pragma unroll
  for (int g = 0; g < 50; ++g)
    do_gate(sr, si, QFC_K[g], QFC_W0[g], QFC_W1[g], rp[g]);
  do_gate(sr, si, 0, 0, 0, qp[0]);                          // RX w0
  do_gate(sr, si, 1, 1, 0, qp[1]);                          // RY w1
  do_gate(sr, si, 2, 3, 0, qp[2]);                          // RZ w3
  do_gate(sr, si, 3, 0, 2, qp[3]);                          // CRX (0->2)
  const float r2 = 0.70710678118654752f;
  rot1 (sr, si, 8 >> 3, r2,0.f,  r2,0.f,   r2,0.f,   -r2,0.f);          // H w3
  rot1 (sr, si, 8 >> 2, .5f,.5f, .5f,-.5f, .5f,-.5f, .5f,.5f);          // SX w2
  crot1(sr, si, 8 >> 3, 8 >> 0, 0.f,0.f, 1.f,0.f, 1.f,0.f, 0.f,0.f);    // CNOT 3->0
  float e[4]; measure_z(sr, si, e);
  #pragma unroll
  for (int w = 0; w < 4; ++w) q[(size_t)b*4 + w] = e[w];
}

// ---------------- K5/K6: BatchNorm1d(4), training mode ---------------------
__global__ __launch_bounds__(256) void k_bnstats(
    const float* __restrict__ q, float* __restrict__ stats) {
  __shared__ float red[8][256];
  const int t = threadIdx.x;
  float s[4] = {0,0,0,0}, ss[4] = {0,0,0,0};
  for (int r = t; r < BATCH; r += 256) {
    #pragma unroll
    for (int c = 0; c < 4; ++c) {
      const float v = q[r*4 + c]; s[c] += v; ss[c] += v*v;
    }
  }
  #pragma unroll
  for (int c = 0; c < 4; ++c) { red[c][t] = s[c]; red[4 + c][t] = ss[c]; }
  __syncthreads();
  for (int off = 128; off > 0; off >>= 1) {
    if (t < off) {
      #pragma unroll
      for (int r = 0; r < 8; ++r) red[r][t] += red[r][t + off];
    }
    __syncthreads();
  }
  if (t < 4) {
    const float mu = red[t][0] * (1.f/BATCH);
    const float ms = red[4 + t][0] * (1.f/BATCH);
    stats[t] = mu;
    stats[4 + t] = ms - mu*mu;                              // biased var
  }
}

__global__ __launch_bounds__(256) void k_bnapply(
    const float* __restrict__ q, const float* __restrict__ stats,
    const float* __restrict__ g, const float* __restrict__ be,
    float* __restrict__ feats) {
  const int tid = blockIdx.x*blockDim.x + threadIdx.x;      // 16384
  const int b = tid >> 2, c = tid & 3;
  const float v = (q[tid] - stats[c]) * rsqrtf(stats[4 + c] + 1e-5f)
                    * g[c] + be[c];
  feats[(size_t)b*68 + 64 + c] = v;
}

// ---------------- K7: FC GEMM on WMMA + log_softmax ------------------------
// M=4096, K=68(pad 96), N=10(pad 16): 256 row tiles, 4 waves/WG x 64 WGs.
__global__ __launch_bounds__(128) void k_fc_logsoftmax(
    const float* __restrict__ feats, const float* __restrict__ fw,
    const float* __restrict__ fb, float* __restrict__ out) {
  __shared__ __align__(16) _Float16 sfe[6144];              // [64 rows][K=96]
  __shared__ __align__(16) _Float16 swf[1536];              // [N=16][K=96]
  __shared__ float slog[1024];                              // 64 rows x 16
  const int t = threadIdx.x, lane = t & 31, wave = t >> 5;
  {                                                         // stage feats tile
    const int row = t >> 1, half = t & 1;
    const size_t rowg = (size_t)blockIdx.x*64 + row;
    #pragma unroll
    for (int u = 0; u < 48; ++u) {
      const int k = half*48 + u;
      const int kc = (k < 68) ? k : 0;
      const float v = feats[rowg*68 + kc];                  // unconditional
      sfe[row*96 + k] = (k < 68) ? (_Float16)v : (_Float16)0.f;
    }
  }
  {                                                         // stage weights
    const int n = t & 15, seg = t >> 4;
    const int nc = (n < 10) ? n : 0;
    #pragma unroll
    for (int u = 0; u < 12; ++u) {
      const int k = seg*12 + u;
      const bool ok = (k < 68) && (n < 10);
      const int kc = (k < 68) ? k : 0;
      const _Float16 v = (_Float16)fw[nc*68 + kc];          // unconditional
      swf[n*96 + k] = ok ? v : (_Float16)0.f;
    }
  }
  __syncthreads();
  const int hi = (lane >> 4) & 1, n = lane & 15;
  const _Float16* ap = &sfe[(wave*16 + (lane & 15))*96];
  const _Float16* bp = &swf[n*96];
  v8f c = {};
  #pragma unroll
  for (int ks = 0; ks < 3; ++ks) {
    const v16h a  = cat8(*(const v8h*)&ap[ks*32 +  8*hi],
                         *(const v8h*)&ap[ks*32 + 16 + 8*hi]);
    const v16h bf = cat8(*(const v8h*)&bp[ks*32 +  8*hi],
                         *(const v8h*)&bp[ks*32 + 16 + 8*hi]);
    c = __builtin_amdgcn_wmma_f32_16x16x32_f16(false, a, false, bf,
                                               (short)0, c, false, false);
  }
  const float bias = (n < 10) ? fb[n] : 0.f;
  #pragma unroll
  for (int r = 0; r < 8; ++r)
    slog[(wave*16 + r + 8*hi)*16 + n] = c[r] + bias;
  __syncthreads();
  if (t < 64) {
    const float* lrow = &slog[t*16];
    float mx = lrow[0];
    #pragma unroll
    for (int j = 1; j < 10; ++j) mx = fmaxf(mx, lrow[j]);
    float se = 0.f;
    #pragma unroll
    for (int j = 0; j < 10; ++j) se += __expf(lrow[j] - mx);
    const float lse = __logf(se);
    const size_t orow = (size_t)blockIdx.x*64 + t;
    #pragma unroll
    for (int j = 0; j < 10; ++j) out[orow*10 + j] = (lrow[j] - mx) - lse;
  }
}

// ---------------------------------------------------------------------------
extern "C" void kernel_launch(void* const* d_in, const int* in_sizes, int n_in,
                              void* d_out, int out_size, void* d_ws, size_t ws_size,
                              hipStream_t stream) {
  const float* x     = (const float*)d_in[0];
  const float* c1w   = (const float*)d_in[1];
  const float* c1b   = (const float*)d_in[2];
  const float* c2w   = (const float*)d_in[3];
  const float* c2b   = (const float*)d_in[4];
  const float* encp  = (const float*)d_in[5];
  const float* qrp   = (const float*)d_in[6];
  const float* qp    = (const float*)d_in[7];
  const float* gamma = (const float*)d_in[8];
  const float* beta  = (const float*)d_in[9];
  const float* fw    = (const float*)d_in[10];
  const float* fb    = (const float*)d_in[11];
  float* out = (float*)d_out;
  (void)in_sizes; (void)n_in; (void)out_size; (void)ws_size;

  char* ws = (char*)d_ws;
  size_t off = 0;
  auto take = [&](size_t bytes) {
    char* p = ws + off;
    off = (off + bytes + 255) & ~(size_t)255;
    return p;
  };
  _Float16* h1   = (_Float16*)take((size_t)BATCH*2048*sizeof(_Float16)); // 16 MB
  float* patches = (float*)take((size_t)BATCH*64*sizeof(float));         // 1 MB
  float* feats   = (float*)take((size_t)BATCH*68*sizeof(float));
  float* st15    = (float*)take((size_t)BATCH*32*sizeof(float));
  float* q       = (float*)take((size_t)BATCH*4*sizeof(float));
  float* stats   = (float*)take(32);

  k_conv1        <<<dim3(BATCH, 8), 256, 0, stream>>>(x, c1w, c1b, h1);
  k_conv2_wmma   <<<BATCH,          128, 0, stream>>>(h1, c2w, c2b, patches);
  k_encoder      <<<256,            256, 0, stream>>>(patches, encp, feats, st15);
  k_qfc          <<<16,             256, 0, stream>>>(st15, qrp, qp, q);
  k_bnstats      <<<1,              256, 0, stream>>>(q, stats);
  k_bnapply      <<<64,             256, 0, stream>>>(q, stats, gamma, beta, feats);
  k_fc_logsoftmax<<<64,             128, 0, stream>>>(feats, fw, fb, out);
}
